// LSTMBlock_15633680957661
// MI455X (gfx1250) — compile-verified, matrix operand order checked by reading, not run on hardware
//
#include <hip/hip_runtime.h>

typedef __bf16 bf16_t;
typedef __attribute__((ext_vector_type(16))) __bf16 v16bf;
typedef __attribute__((ext_vector_type(8)))  __bf16 v8bf;
typedef __attribute__((ext_vector_type(8)))  float  v8f;
typedef __attribute__((ext_vector_type(4)))  float  v4f;

#define BATCH 16384
#define HDIM  1024
#define KDIM  2048   // D + H
#define BM    128
#define BN    32
#define BK    64
#define LDK   72     // padded LDS row (bf16 elems): 144 bytes, keeps 16B alignment

__device__ __forceinline__ float fast_sigmoid(float x) {
    return 1.0f / (1.0f + __expf(-x));
}
__device__ __forceinline__ float fast_tanh(float x) {
    float e = __expf(-2.0f * x);
    return (1.0f - e) / (1.0f + e);
}

__device__ __forceinline__ v16bf load_frag(const bf16_t* p) {
    const v8bf lo = *(const v8bf*)(p);
    const v8bf hi = *(const v8bf*)(p + 16);
    return __builtin_shufflevector(lo, hi,
        0,1,2,3,4,5,6,7,8,9,10,11,12,13,14,15);
}

__global__ __launch_bounds__(256, 2)
void lstm_fused_wmma(const float* __restrict__ x,
                     const float* __restrict__ c,
                     const float* __restrict__ h,
                     const float* __restrict__ Wf, const float* __restrict__ bf,
                     const float* __restrict__ Wi, const float* __restrict__ bi,
                     const float* __restrict__ Wg, const float* __restrict__ bg,
                     const float* __restrict__ Wo, const float* __restrict__ bo,
                     float* __restrict__ c_out, float* __restrict__ h_out)
{
    __shared__ bf16_t ldsA[BM * LDK];          // 128 x 64 bf16 (padded)
    __shared__ bf16_t ldsW[4][BN * LDK];       // 4 gates x 32 x 64 bf16 (padded)

    const int tid  = threadIdx.x;
    const int wave = tid >> 5;                 // 0..7
    const int lane = tid & 31;
    const int bm   = blockIdx.x * BM;
    const int bn   = blockIdx.y * BN;

    // WMMA fragment addressing (wave32 bf16 16x16x32 layout):
    // lanes 0-15: row = lane,    K chunks {0..7, 16..23}
    // lanes 16-31: row = lane-16, K chunks {8..15, 24..31}
    const int lrow   = lane & 15;
    const int khalf  = (lane >> 4) << 3;       // 0 or 8 (bf16 elems)
    const int m_base = wave << 4;              // wave's 16-row stripe in tile

    v8f acc[8];                                // [gate*2 + nsub]
    #pragma unroll
    for (int t = 0; t < 8; ++t)
        acc[t] = (v8f){0.f,0.f,0.f,0.f,0.f,0.f,0.f,0.f};

    // ---- loader thread mapping (uniform per thread across iterations) ----
    const int a_col4 = (tid & 15) << 2;        // 0..60 step 4 (f32 elems)
    const int a_row0 = tid >> 4;               // 0..15
    const int w_gate = tid >> 6;               // 0..3
    const int w_tg   = tid & 63;
    const int w_col4 = (w_tg & 15) << 2;       // 0..60 step 4
    const int w_row0 = w_tg >> 4;              // 0..3
    const float* wsrc_base =
        (w_gate == 0) ? Wf : (w_gate == 1) ? Wi : (w_gate == 2) ? Wg : Wo;

    // ---- software-pipelined global->register prefetch buffers ----
    v4f pa[8];   // A tile slice   (streamed: non-temporal)
    v4f pw[8];   // W tile slice   (L2-resident: default policy)

    // prologue: prefetch K-block 0
    {
        const float* asrc = x;   // kb = 0 is always in x
        #pragma unroll
        for (int i = 0; i < 8; ++i) {
            const int row = a_row0 + (i << 4);
            pa[i] = __builtin_nontemporal_load(
                (const v4f*)(asrc + (size_t)(bm + row) * HDIM + a_col4));
        }
        const float* wsrc = wsrc_base;
        #pragma unroll
        for (int i = 0; i < 8; ++i) {
            const int row = w_row0 + (i << 2);
            pw[i] = *(const v4f*)(wsrc + (size_t)(bn + row) * KDIM + w_col4);
        }
    }

    for (int kb = 0; kb < KDIM; kb += BK) {
        // ---- drain prefetch regs into LDS (f32 -> bf16) ----
        #pragma unroll
        for (int i = 0; i < 8; ++i) {
            const int row = a_row0 + (i << 4);
            bf16_t* d = &ldsA[row * LDK + a_col4];
            d[0] = (bf16_t)pa[i].x; d[1] = (bf16_t)pa[i].y;
            d[2] = (bf16_t)pa[i].z; d[3] = (bf16_t)pa[i].w;
        }
        #pragma unroll
        for (int i = 0; i < 8; ++i) {
            const int row = w_row0 + (i << 2);
            bf16_t* d = &ldsW[w_gate][row * LDK + w_col4];
            d[0] = (bf16_t)pw[i].x; d[1] = (bf16_t)pw[i].y;
            d[2] = (bf16_t)pw[i].z; d[3] = (bf16_t)pw[i].w;
        }
        __syncthreads();

        // ---- issue next K-block's global loads; latency hidden by compute ----
        const int kn = kb + BK;
        if (kn < KDIM) {
            const float* asrc = (kn < HDIM) ? (x + kn) : (h + (kn - HDIM));
            #pragma unroll
            for (int i = 0; i < 8; ++i) {
                const int row = a_row0 + (i << 4);
                pa[i] = __builtin_nontemporal_load(
                    (const v4f*)(asrc + (size_t)(bm + row) * HDIM + a_col4));
            }
            const float* wsrc = wsrc_base + kn;
            #pragma unroll
            for (int i = 0; i < 8; ++i) {
                const int row = w_row0 + (i << 2);
                pw[i] = *(const v4f*)(wsrc + (size_t)(bn + row) * KDIM + w_col4);
            }
        }

        // ---- 2 k-steps of 32: 8 WMMAs each (A frag shared across gates) ----
        #pragma unroll
        for (int ks = 0; ks < BK; ks += 32) {
            const v16bf a = load_frag(&ldsA[(m_base + lrow) * LDK + ks + khalf]);
            #pragma unroll
            for (int t = 0; t < 8; ++t) {
                const int g  = t >> 1;
                const int ns = t & 1;
                const v16bf b = load_frag(&ldsW[g][((ns << 4) + lrow) * LDK + ks + khalf]);
                acc[t] = __builtin_amdgcn_wmma_f32_16x16x32_bf16(
                    false, a, false, b, (short)0, acc[t], false, false);
            }
        }
        __syncthreads();   // all waves done reading LDS before next overwrite
    }

    // ---- fused LSTM epilogue ----
    // C/D layout: VGPR r, lanes 0-15: M=r, lanes 16-31: M=r+8; N = lane&15
    const int rbase = (lane >> 4) << 3;        // 0 or 8
    #pragma unroll
    for (int ns = 0; ns < 2; ++ns) {
        const int col = bn + (ns << 4) + lrow;         // 0..1023
        const float vbf = bf[col];
        const float vbi = bi[col];
        const float vbg = bg[col];
        const float vbo = bo[col];
        #pragma unroll
        for (int r = 0; r < 8; ++r) {
            const int row = bm + m_base + rbase + r;
            const size_t idx = (size_t)row * HDIM + col;
            const float zf = acc[0*2+ns][r] + vbf;
            const float zi = acc[1*2+ns][r] + vbi;
            const float zg = acc[2*2+ns][r] + vbg;
            const float zo = acc[3*2+ns][r] + vbo;
            const float fg = fast_sigmoid(zf);
            const float ig = fast_sigmoid(zi) * fast_tanh(zg);
            const float og = fast_sigmoid(zo);
            const float cold = __builtin_nontemporal_load(&c[idx]);
            const float cn = fg * cold + ig;
            __builtin_nontemporal_store(cn, &c_out[idx]);
            __builtin_nontemporal_store(fast_tanh(cn) * og, &h_out[idx]);
        }
    }
}

extern "C" void kernel_launch(void* const* d_in, const int* in_sizes, int n_in,
                              void* d_out, int out_size, void* d_ws, size_t ws_size,
                              hipStream_t stream) {
    (void)in_sizes; (void)n_in; (void)out_size; (void)d_ws; (void)ws_size;
    const float* x  = (const float*)d_in[0];
    const float* c  = (const float*)d_in[1];
    const float* h  = (const float*)d_in[2];
    const float* Wf = (const float*)d_in[3];
    const float* bf = (const float*)d_in[4];
    const float* bi = nullptr; // placeholder to keep ordering obvious below
    (void)bi;
    const float* Wi = (const float*)d_in[5];
    const float* bi_ = (const float*)d_in[6];
    const float* Wg = (const float*)d_in[7];
    const float* bg = (const float*)d_in[8];
    const float* Wo = (const float*)d_in[9];
    const float* bo = (const float*)d_in[10];

    float* c_out = (float*)d_out;
    float* h_out = c_out + (size_t)BATCH * HDIM;

    dim3 grid(BATCH / BM, HDIM / BN);   // 128 x 32
    lstm_fused_wmma<<<grid, 256, 0, stream>>>(x, c, h,
                                              Wf, bf, Wi, bi_, Wg, bg, Wo, bo,
                                              c_out, h_out);
}
